// GenericCrossTransformerLayer_4552665334364
// MI455X (gfx1250) — matrix operand order repaired and checked by reading, not compiled, not run on hardware
//
#include <hip/hip_runtime.h>

// ---------------------------------------------------------------------------
// CDNA5 (gfx1250) cross-transformer layer, bf16 WMMA pipeline.
// All matmuls use v_wmma_f32_16x16x32_bf16 (f32 accumulate).
// ---------------------------------------------------------------------------

typedef __attribute__((ext_vector_type(16))) __bf16 v16bf;
typedef __attribute__((ext_vector_type(8)))  float  v8f;

union FragA { v16bf v; uint4 q[2]; unsigned u[8]; };

#define WMMA_BF16(a, b, c) \
  __builtin_amdgcn_wmma_f32_16x16x32_bf16(false, (a), false, (b), (short)0, (c), false, false)

#define BB   2
#define LL   2048
#define TT   4096      // B*L tokens
#define DM   768
#define HH   12
#define DH   64
#define DFF  2304

// ---------------------------------------------------------------- fp32->bf16
__global__ __launch_bounds__(256) void k_cvt(const float* __restrict__ s,
                                             __bf16* __restrict__ d, int n) {
  int i = blockIdx.x * 256 + threadIdx.x;
  if (i < n) d[i] = (__bf16)s[i];
}

// ---------------------------------------------------------------- RMS norm
__global__ __launch_bounds__(256) void k_rms(const float* __restrict__ x,
                                             const float* __restrict__ g,
                                             __bf16* __restrict__ out, int D) {
  __shared__ float red[8];
  const size_t row = blockIdx.x;
  const float* xr = x + row * D;
  float ss = 0.f;
  for (int i = threadIdx.x; i < D; i += 256) { float v = xr[i]; ss += v * v; }
#pragma unroll
  for (int m = 16; m; m >>= 1) ss += __shfl_xor(ss, m, 32);
  if ((threadIdx.x & 31) == 0) red[threadIdx.x >> 5] = ss;
  __syncthreads();
  float tot = 0.f;
#pragma unroll
  for (int w = 0; w < 8; ++w) tot += red[w];
  const float inv = rsqrtf(tot / (float)D + 1e-6f);
  for (int i = threadIdx.x; i < D; i += 256)
    out[row * D + i] = (__bf16)(xr[i] * g[i] * inv);
}

// ---------------------------------------------------------------- WMMA GEMM
// C[M,N] = A[M,K] * W[N,K]^T (+ resid).  A,W bf16 row-major, C f32.
// Wave tile 32x64, block tile 64x256 (8 waves).  M%64==0, N%256==0, K%32==0.
__global__ __launch_bounds__(256) void k_gemm(const __bf16* __restrict__ A,
                                              const __bf16* __restrict__ W,
                                              const float* __restrict__ R,
                                              float* __restrict__ C,
                                              int M, int N, int K) {
  const int wv = threadIdx.x >> 5, lane = threadIdx.x & 31;
  const int half = lane >> 4, l16 = lane & 15;
  const int nbn = N >> 8;
  const int bm = blockIdx.x / nbn, bn = blockIdx.x % nbn;
  const int m0 = bm * 64 + (wv >> 2) * 32;
  const int n0 = bn * 256 + (wv & 3) * 64;

  v8f acc[2][4] = {};
  const __bf16* Arow0 = A + (size_t)(m0 + l16) * K;
  const __bf16* Arow1 = A + (size_t)(m0 + 16 + l16) * K;
  const __bf16* Wr[4];
#pragma unroll
  for (int nt = 0; nt < 4; ++nt) Wr[nt] = W + (size_t)(n0 + nt * 16 + l16) * K;

  for (int k0 = 0; k0 < K; k0 += 32) {
    FragA a[2], b[4];
    const int ab = k0 + half * 8;   // A frag: lanes 0-15 K{0..7,16..23}, 16-31 K{8..15,24..31}
    a[0].q[0] = *(const uint4*)(Arow0 + ab);
    a[0].q[1] = *(const uint4*)(Arow0 + ab + 16);
    a[1].q[0] = *(const uint4*)(Arow1 + ab);
    a[1].q[1] = *(const uint4*)(Arow1 + ab + 16);
    const int bb = k0 + half * 8;   // B frag: lanes 0-15 K{0..7,16..23}, 16-31 K{8..15,24..31}
#pragma unroll
    for (int nt = 0; nt < 4; ++nt) {
      b[nt].q[0] = *(const uint4*)(Wr[nt] + bb);
      b[nt].q[1] = *(const uint4*)(Wr[nt] + bb + 16);
    }
    __builtin_prefetch(Arow0 + ab + 64, 0, 0);  // global_prefetch_b8
#pragma unroll
    for (int mt = 0; mt < 2; ++mt)
#pragma unroll
      for (int nt = 0; nt < 4; ++nt)
        acc[mt][nt] = WMMA_BF16(a[mt].v, b[nt].v, acc[mt][nt]);
  }

#pragma unroll
  for (int mt = 0; mt < 2; ++mt)
#pragma unroll
    for (int nt = 0; nt < 4; ++nt) {
      const int col = n0 + nt * 16 + l16;
#pragma unroll
      for (int j = 0; j < 8; ++j) {
        const int row = m0 + mt * 16 + half * 8 + j;  // C layout: VGPR j -> M=j+8*half
        const size_t idx = (size_t)row * N + col;
        float v = acc[mt][nt][j];
        if (R) v += R[idx];
        C[idx] = v;
      }
    }
}

// --------------------------------------------------- cos-scale + RoPE prep
// Emits Q,K as [B,H,L,64] bf16 (A/B fragment friendly) and V^T as [B,H,64,L].
__global__ __launch_bounds__(256) void k_attn_prep(
    const float* __restrict__ qsrc, int qstr, int qoff,
    const float* __restrict__ kvsrc, int kvstr, int koff, int voff,
    const float* __restrict__ posq, const float* __restrict__ posk,
    const float* __restrict__ scale,
    __bf16* __restrict__ qh, __bf16* __restrict__ kh, __bf16* __restrict__ vt,
    int L, int Hn) {
  __shared__ float S[8][2][64];
  const int wv = threadIdx.x >> 5, lane = threadIdx.x & 31;
  const long r = (long)blockIdx.x * 8 + wv;     // (b*H + h)*L + l
  const int l = (int)(r % L);
  const int bh = (int)(r / L);
  const int h = bh % Hn, b = bh / Hn;
  const long tq = (long)b * L + l;
  const int d0 = lane, d1 = lane + 32;

  const float* qp = qsrc + tq * qstr + qoff + h * 64;
  const float* kp = kvsrc + tq * kvstr + koff + h * 64;
  const float* vp = kvsrc + tq * kvstr + voff + h * 64;
  float q0 = qp[d0], q1 = qp[d1];
  float k0 = kp[d0], k1 = kp[d1];
  float v0 = vp[d0], v1 = vp[d1];
  float sq = q0 * q0 + q1 * q1, sk = k0 * k0 + k1 * k1;
#pragma unroll
  for (int m = 16; m; m >>= 1) { sq += __shfl_xor(sq, m, 32); sk += __shfl_xor(sk, m, 32); }
  const float s = sqrtf(scale[h]);
  const float qs = s * rsqrtf(sq + 1e-6f), ks = s * rsqrtf(sk + 1e-6f);
  S[wv][0][d0] = q0 * qs; S[wv][0][d1] = q1 * qs;
  S[wv][1][d0] = k0 * ks; S[wv][1][d1] = k1 * ks;
  __syncthreads();

  const float* pq = posq + tq * 3;
  const float* pk = posk + tq * 3;
  auto rope = [&](const float* Sr, const float* pos, int d) -> float {
    if (d >= 48) return Sr[d];
    const int j = (d < 24) ? d : d - 24;
    const int c = j >> 3, f = j & 7;
    // freq = pi * 10^((f*12 + h)/96)
    const float freq = __expf(1.14472988584940017f + (float)(f * 12 + h) * 0.02398526134535464f);
    const float th = pos[c] * freq;
    const float cs = __cosf(th), sn = __sinf(th);
    return (d < 24) ? (Sr[d] * cs - Sr[d + 24] * sn) : (Sr[d] * cs + Sr[d - 24] * sn);
  };
  const long orow = ((long)bh * L + l) * 64;
  qh[orow + d0] = (__bf16)rope(S[wv][0], pq, d0);
  qh[orow + d1] = (__bf16)rope(S[wv][0], pq, d1);
  kh[orow + d0] = (__bf16)rope(S[wv][1], pk, d0);
  kh[orow + d1] = (__bf16)rope(S[wv][1], pk, d1);
  const long vbase = ((long)bh * 64) * L + l;
  vt[vbase + (long)d0 * L] = (__bf16)v0;
  vt[vbase + (long)d1 * L] = (__bf16)v1;
}

// ----------------------------------------------------- flash attention core
// One wave per 32-query tile; online softmax; S/P staged through LDS.
__global__ __launch_bounds__(256) void k_attn(const __bf16* __restrict__ qhp,
                                              const __bf16* __restrict__ khp,
                                              const __bf16* __restrict__ vtp,
                                              __bf16* __restrict__ o,
                                              int Hn, int L) {
  __shared__ float  Sb[8][32][33];
  __shared__ __bf16 Pb[8][32][32];
  const int wv = threadIdx.x >> 5, lane = threadIdx.x & 31;
  const int half = lane >> 4, l16 = lane & 15;
  const int ntq = L >> 5;
  const int tile = blockIdx.x * 8 + wv;
  const int bh = tile / ntq;
  const int qbase = (tile % ntq) * 32;
  const int b = bh / Hn, h = bh % Hn;
  const __bf16* Q  = qhp + (size_t)bh * L * 64;
  const __bf16* Kp = khp + (size_t)bh * L * 64;
  const __bf16* Vt = vtp + (size_t)bh * 64 * L;

  FragA qa[2][2];  // [m-tile][k-step of 32 over head dim]
#pragma unroll
  for (int mt = 0; mt < 2; ++mt)
#pragma unroll
    for (int ks = 0; ks < 2; ++ks) {
      const __bf16* r = Q + (size_t)(qbase + mt * 16 + l16) * 64 + ks * 32 + half * 8;
      qa[mt][ks].q[0] = *(const uint4*)r;
      qa[mt][ks].q[1] = *(const uint4*)(r + 16);
    }

  float mrow = -1e30f, lrow = 0.f;   // lane == query row (0..31)
  v8f oacc[2][4] = {};

  for (int key0 = 0; key0 < L; key0 += 32) {
    // ---- S = Q * K^T for this 32-key chunk
    v8f sacc[2][2] = {};
    FragA kb[2][2];
#pragma unroll
    for (int nt = 0; nt < 2; ++nt)
#pragma unroll
      for (int ks = 0; ks < 2; ++ks) {
        const __bf16* r = Kp + (size_t)(key0 + nt * 16 + l16) * 64 + ks * 32 + half * 8;
        kb[nt][ks].q[0] = *(const uint4*)r;
        kb[nt][ks].q[1] = *(const uint4*)(r + 16);
      }
#pragma unroll
    for (int mt = 0; mt < 2; ++mt)
#pragma unroll
      for (int nt = 0; nt < 2; ++nt)
#pragma unroll
        for (int ks = 0; ks < 2; ++ks)
          sacc[mt][nt] = WMMA_BF16(qa[mt][ks].v, kb[nt][ks].v, sacc[mt][nt]);

    // ---- C-layout -> LDS [query][key]
#pragma unroll
    for (int mt = 0; mt < 2; ++mt)
#pragma unroll
      for (int nt = 0; nt < 2; ++nt)
#pragma unroll
        for (int j = 0; j < 8; ++j)
          Sb[wv][mt * 16 + half * 8 + j][nt * 16 + l16] = sacc[mt][nt][j];

    // ---- online softmax: each lane owns one query row
    float* Sr = Sb[wv][lane];
    float cm = -1e30f;
#pragma unroll
    for (int j = 0; j < 32; ++j) cm = fmaxf(cm, Sr[j]);
    const float mnew = fmaxf(mrow, cm);
    const float corr = __expf(mrow - mnew);
    float psum = 0.f;
#pragma unroll
    for (int j = 0; j < 32; ++j) {
      const float p = __expf(Sr[j] - mnew);
      psum += p;
      Pb[wv][lane][j] = (__bf16)p;
    }
    lrow = lrow * corr + psum;
    mrow = mnew;

    // ---- rescale O accumulators by per-row correction (via shfl from lane==row)
#pragma unroll
    for (int mt = 0; mt < 2; ++mt)
#pragma unroll
      for (int j = 0; j < 8; ++j) {
        const float cj = __shfl(corr, mt * 16 + half * 8 + j, 32);
#pragma unroll
        for (int nt = 0; nt < 4; ++nt) oacc[mt][nt][j] *= cj;
      }

    // ---- P fragments (A layout) from LDS
    FragA pa[2];
#pragma unroll
    for (int mt = 0; mt < 2; ++mt) {
      const unsigned* Pr = (const unsigned*)(&Pb[wv][mt * 16 + l16][0]);
      const int kb4 = half * 4;
#pragma unroll
      for (int t = 0; t < 4; ++t) {
        pa[mt].u[t]     = Pr[kb4 + t];
        pa[mt].u[4 + t] = Pr[8 + kb4 + t];
      }
    }
    // ---- V fragments (B layout) from transposed V: contiguous along keys
    FragA vb[4];
#pragma unroll
    for (int nt = 0; nt < 4; ++nt) {
      const __bf16* r = Vt + (size_t)(nt * 16 + l16) * L + key0 + half * 8;
      vb[nt].q[0] = *(const uint4*)r;
      vb[nt].q[1] = *(const uint4*)(r + 16);
    }
#pragma unroll
    for (int mt = 0; mt < 2; ++mt)
#pragma unroll
      for (int nt = 0; nt < 4; ++nt)
        oacc[mt][nt] = WMMA_BF16(pa[mt].v, vb[nt].v, oacc[mt][nt]);
  }

  // ---- final 1/l normalization + store [B, L, H*64] bf16
  const float linv = 1.f / lrow;
#pragma unroll
  for (int mt = 0; mt < 2; ++mt)
#pragma unroll
    for (int j = 0; j < 8; ++j) {
      const float lj = __shfl(linv, mt * 16 + half * 8 + j, 32);
      const int qi = qbase + mt * 16 + half * 8 + j;
#pragma unroll
      for (int nt = 0; nt < 4; ++nt) {
        const int dim = nt * 16 + l16;
        o[((size_t)b * L + qi) * (size_t)(Hn * 64) + h * 64 + dim] =
            (__bf16)(oacc[mt][nt][j] * lj);
      }
    }
}

// ---------------------------------------------------------------- SwiGLU
__global__ __launch_bounds__(256) void k_swiglu(const float* __restrict__ u,
                                                __bf16* __restrict__ out,
                                                int T, int F) {
  const int i = blockIdx.x * 256 + threadIdx.x;
  if (i >= T * F) return;
  const int t = i / F, j = i % F;
  const float a = u[(size_t)t * 2 * F + j];
  const float g = u[(size_t)t * 2 * F + F + j];
  const float sig = 1.f / (1.f + __expf(-g));
  out[i] = (__bf16)(a * g * sig);
}

// ---------------------------------------------------------------------------
extern "C" void kernel_launch(void* const* d_in, const int* in_sizes, int n_in,
                              void* d_out, int out_size, void* d_ws, size_t ws_size,
                              hipStream_t stream) {
  const float* x         = (const float*)d_in[0];
  const float* pos       = (const float*)d_in[1];
  const float* x_cross   = (const float*)d_in[2];
  const float* pos_cross = (const float*)d_in[3];
  const float* sa_norm   = (const float*)d_in[4];
  const float* sa_wqkv   = (const float*)d_in[5];
  const float* sa_scale  = (const float*)d_in[6];
  const float* sa_wout   = (const float*)d_in[7];
  const float* ca_norm   = (const float*)d_in[8];
  const float* ca_normc  = (const float*)d_in[9];
  const float* ca_wq     = (const float*)d_in[10];
  const float* ca_wkv    = (const float*)d_in[11];
  const float* ca_scale  = (const float*)d_in[12];
  const float* ca_wout   = (const float*)d_in[13];
  const float* ff_norm   = (const float*)d_in[14];
  const float* ff_wup    = (const float*)d_in[15];
  const float* ff_wdown  = (const float*)d_in[16];
  float* out = (float*)d_out;
  (void)in_sizes; (void)n_in; (void)out_size; (void)ws_size;

  char* base = (char*)d_ws;
  size_t off = 0;
  auto take = [&](size_t bytes) -> char* {
    char* p = base + off;
    off = (off + bytes + 255) & ~(size_t)255;
    return p;
  };
  __bf16* wqkv_b  = (__bf16*)take((size_t)3 * DM * DM * 2);
  __bf16* wsao_b  = (__bf16*)take((size_t)DM * DM * 2);
  __bf16* wcaq_b  = (__bf16*)take((size_t)DM * DM * 2);
  __bf16* wcakv_b = (__bf16*)take((size_t)2 * DM * DM * 2);
  __bf16* wcao_b  = (__bf16*)take((size_t)DM * DM * 2);
  __bf16* wup_b   = (__bf16*)take((size_t)2 * DFF * DM * 2);
  __bf16* wdn_b   = (__bf16*)take((size_t)DM * DFF * 2);
  __bf16* h_b     = (__bf16*)take((size_t)TT * DM * 2);
  float*  big     = (float*) take((size_t)TT * 2 * DFF * 4);  // qkv / kv / up-proj
  float*  qf      = (float*) take((size_t)TT * DM * 4);
  __bf16* qh      = (__bf16*)take((size_t)TT * DM * 2);
  __bf16* kh      = (__bf16*)take((size_t)TT * DM * 2);
  __bf16* vt      = (__bf16*)take((size_t)TT * DM * 2);
  __bf16* o_b     = (__bf16*)take((size_t)TT * DM * 2);
  float*  xa      = (float*) take((size_t)TT * DM * 4);
  __bf16* swi     = (__bf16*)take((size_t)TT * DFF * 2);

  auto cvt = [&](const float* s, __bf16* d, size_t n) {
    k_cvt<<<(unsigned)((n + 255) / 256), 256, 0, stream>>>(s, d, (int)n);
  };
  auto gemm = [&](const __bf16* A, const __bf16* W, const float* R, float* C,
                  int M, int N, int K) {
    k_gemm<<<(unsigned)((M / 64) * (N / 256)), 256, 0, stream>>>(A, W, R, C, M, N, K);
  };

  // weights -> bf16
  cvt(sa_wqkv, wqkv_b,  (size_t)3 * DM * DM);
  cvt(sa_wout, wsao_b,  (size_t)DM * DM);
  cvt(ca_wq,   wcaq_b,  (size_t)DM * DM);
  cvt(ca_wkv,  wcakv_b, (size_t)2 * DM * DM);
  cvt(ca_wout, wcao_b,  (size_t)DM * DM);
  cvt(ff_wup,  wup_b,   (size_t)2 * DFF * DM);
  cvt(ff_wdown,wdn_b,   (size_t)DM * DFF);

  const unsigned prep_grid = (TT * HH) / 8;           // 6144
  const unsigned attn_grid = (BB * HH * (LL / 32)) / 8; // 192

  // ---- self attention
  k_rms<<<TT, 256, 0, stream>>>(x, sa_norm, h_b, DM);
  gemm(h_b, wqkv_b, nullptr, big, TT, 3 * DM, DM);
  k_attn_prep<<<prep_grid, 256, 0, stream>>>(big, 3 * DM, 0, big, 3 * DM, DM, 2 * DM,
                                             pos, pos, sa_scale, qh, kh, vt, LL, HH);
  k_attn<<<attn_grid, 256, 0, stream>>>(qh, kh, vt, o_b, HH, LL);
  gemm(o_b, wsao_b, x, xa, TT, DM, DM);

  // ---- cross attention
  k_rms<<<TT, 256, 0, stream>>>(xa, ca_norm, h_b, DM);
  gemm(h_b, wcaq_b, nullptr, qf, TT, DM, DM);
  k_rms<<<TT, 256, 0, stream>>>(x_cross, ca_normc, h_b, DM);
  gemm(h_b, wcakv_b, nullptr, big, TT, 2 * DM, DM);
  k_attn_prep<<<prep_grid, 256, 0, stream>>>(qf, DM, 0, big, 2 * DM, 0, DM,
                                             pos, pos_cross, ca_scale, qh, kh, vt, LL, HH);
  k_attn<<<attn_grid, 256, 0, stream>>>(qh, kh, vt, o_b, HH, LL);
  gemm(o_b, wcao_b, xa, xa, TT, DM, DM);

  // ---- SwiGLU FFN
  k_rms<<<TT, 256, 0, stream>>>(xa, ff_norm, h_b, DM);
  gemm(h_b, wup_b, nullptr, big, TT, 2 * DFF, DM);
  k_swiglu<<<(unsigned)(((size_t)TT * DFF + 255) / 256), 256, 0, stream>>>(big, swi, TT, DFF);
  gemm(swi, wdn_b, xa, out, TT, DM, DFF);
}
